// WindowAttention3D_43241730736492
// MI455X (gfx1250) — matrix operand order repaired and checked by reading, not compiled, not run on hardware
//
#include <hip/hip_runtime.h>
#include <hip/hip_bf16.h>

// ---------------------------------------------------------------------------
// WindowAttention3D forward for gfx1250 (CDNA5, wave32, WMMA).
// Shapes: B=1024, N1=256, N2=128, C=128, NH=4, hd=32, NW=64.
// Outputs (concat): x (B,N1,C) f32  then attn (B,NH,N1,N2) f32.
// ---------------------------------------------------------------------------

typedef __attribute__((ext_vector_type(16))) _Float16 v16h;
typedef __attribute__((ext_vector_type(8)))  _Float16 v8h;
typedef __attribute__((ext_vector_type(8)))  float    v8f;
typedef __attribute__((ext_vector_type(4)))  float    v4f;

#define DEVFN static __device__ __forceinline__

// ---- WMMA wrapper: D = A(16x32 f16) * B(32x16 f16) + C(16x16 f32) ---------
DEVFN v8f wmma16(v16h a, v16h b, v8f c) {
  return __builtin_amdgcn_wmma_f32_16x16x32_f16(
      /*neg_a=*/false, a, /*neg_b=*/false, b,
      /*c_mod=*/(short)0, c, /*reuse_a=*/false, /*reuse_b=*/false);
}

// ---- A fragment (16x32 f16, row-major source, stride ld elements) ----------
// ISA 7.12.2: lanes 0-15 hold row m=lane, K = {0..7, 16..23};
//             lanes 16-31 hold row m=lane-16, K = {8..15, 24..31}.
DEVFN v16h a_frag_f16(const _Float16* __restrict__ base, int ld, int lane) {
  const int m  = lane & 15;
  const int c0 = (lane & 16) ? 8 : 0;
  const _Float16* p = base + m * ld + c0;
  v8h lo = *(const v8h*)(p);        // K = c0 .. c0+7      (16B aligned)
  v8h hi = *(const v8h*)(p + 16);   // K = c0+16 .. c0+23
  v16h r;
#pragma unroll
  for (int i = 0; i < 8; ++i) { r[i] = lo[i]; r[i + 8] = hi[i]; }
  return r;
}

// Same fragment but source is f32 (converted on the fly).
DEVFN v16h a_frag_f32(const float* __restrict__ base, int ld, int lane) {
  const int m  = lane & 15;
  const int c0 = (lane & 16) ? 8 : 0;
  const float* p = base + m * ld + c0;
  v4f x0 = *(const v4f*)(p);
  v4f x1 = *(const v4f*)(p + 4);
  v4f x2 = *(const v4f*)(p + 16);
  v4f x3 = *(const v4f*)(p + 20);
  v16h r;
#pragma unroll
  for (int i = 0; i < 4; ++i) {
    r[i]      = (_Float16)x0[i];
    r[i + 4]  = (_Float16)x1[i];
    r[i + 8]  = (_Float16)x2[i];
    r[i + 12] = (_Float16)x3[i];
  }
  return r;
}

// ---- B fragment (32x16 f16). Source layout: for each output column n,
// the 32 K-values are contiguous (stride ld elements between columns).
// ISA: lanes 0-15 -> col n=lane, K={0..7,16..23}; lanes 16-31 -> col n=lane-16, K={8..15,24..31}.
DEVFN v16h b_frag_f16(const _Float16* __restrict__ base, int ld, int lane) {
  const int n  = lane & 15;
  const int kb = (lane & 16) ? 8 : 0;
  const _Float16* p = base + n * ld + kb;
  v8h lo = *(const v8h*)(p);
  v8h hi = *(const v8h*)(p + 16);
  v16h r;
#pragma unroll
  for (int i = 0; i < 8; ++i) { r[i] = lo[i]; r[i + 8] = hi[i]; }
  return r;
}

// ---------------------------------------------------------------------------
// prep: weights -> f16, bias table bias[h][n1][n2] = rpb[rel_index[n1,n2], h]
// ---------------------------------------------------------------------------
__global__ __launch_bounds__(256) void prep_kernel(
    const float* __restrict__ qw, const float* __restrict__ kvw,
    const float* __restrict__ pw, const float* __restrict__ rpb,
    const int* __restrict__ ridx,
    _Float16* __restrict__ qwh, _Float16* __restrict__ kvwh,
    _Float16* __restrict__ pwh, float* __restrict__ biasT) {
  const int i = blockIdx.x * 256 + threadIdx.x;   // grid covers 131072
  if (i < 128 * 128) pwh[i] = (_Float16)pw[i];
  if (i < 128 * 128) qwh[i] = (_Float16)qw[i];
  if (i < 256 * 128) kvwh[i] = (_Float16)kvw[i];
  if (i < 4 * 256 * 128) {
    const int h = i >> 15;          // 0..3
    const int r = i & 32767;        // n1*128+n2
    biasT[h * 32768 + r] = rpb[ridx[r] * 4 + h];
  }
}

// ---------------------------------------------------------------------------
// qproj: qh = (q @ q_w^T + q_b) * scale, f16 out, layout (B*N1, 128)
// C starts as inline 0 (no VGPR init); bias folded into epilogue.
// ---------------------------------------------------------------------------
__global__ __launch_bounds__(256) void qproj_kernel(
    const float* __restrict__ q, const _Float16* __restrict__ wh,
    const float* __restrict__ qb, _Float16* __restrict__ qh) {
  const int lane = threadIdx.x & 31;
  const int wave = threadIdx.x >> 5;
  const int row0 = (blockIdx.x * 8 + wave) * 16;   // 16384 m-tiles
  const float scale = 0.17677669529663687f;        // hd^-0.5

  v16h a[4];
#pragma unroll
  for (int ks = 0; ks < 4; ++ks)
    a[ks] = a_frag_f32(q + row0 * 128 + ks * 32, 128, lane);

  const int col16 = lane & 15;
  const int rAdd  = (lane >> 4) << 3;
#pragma unroll
  for (int nt = 0; nt < 8; ++nt) {
    v8f c = {};
#pragma unroll
    for (int ks = 0; ks < 4; ++ks) {
      v16h b = b_frag_f16(wh + (nt * 16) * 128 + ks * 32, 128, lane);
      c = wmma16(a[ks], b, c);
    }
    const float bv = qb[nt * 16 + col16];
    const int col = nt * 16 + col16;
#pragma unroll
    for (int r = 0; r < 8; ++r)
      qh[(row0 + rAdd + r) * 128 + col] = (_Float16)((c[r] + bv) * scale);
  }
}

// ---------------------------------------------------------------------------
// kvproj: (kv @ kv_w^T + kv_b) -> k (features 0..127), v (features 128..255)
// ---------------------------------------------------------------------------
__global__ __launch_bounds__(256) void kvproj_kernel(
    const float* __restrict__ kv, const _Float16* __restrict__ wh,
    const float* __restrict__ kvb, _Float16* __restrict__ kh,
    _Float16* __restrict__ vh) {
  const int lane = threadIdx.x & 31;
  const int wave = threadIdx.x >> 5;
  const int row0 = (blockIdx.x * 8 + wave) * 16;   // 8192 m-tiles

  v16h a[4];
#pragma unroll
  for (int ks = 0; ks < 4; ++ks)
    a[ks] = a_frag_f32(kv + row0 * 128 + ks * 32, 128, lane);

  const int col16 = lane & 15;
  const int rAdd  = (lane >> 4) << 3;
#pragma unroll
  for (int nt = 0; nt < 16; ++nt) {
    v8f c = {};
#pragma unroll
    for (int ks = 0; ks < 4; ++ks) {
      v16h b = b_frag_f16(wh + (nt * 16) * 128 + ks * 32, 128, lane);
      c = wmma16(a[ks], b, c);
    }
    const float bv = kvb[nt * 16 + col16];
    const int j = nt * 16 + col16;                 // 0..255
    _Float16* __restrict__ dst = (j < 128) ? kh : vh;
    const int jj = j & 127;
#pragma unroll
    for (int r = 0; r < 8; ++r)
      dst[(row0 + rAdd + r) * 128 + jj] = (_Float16)(c[r] + bv);
  }
}

// ---------------------------------------------------------------------------
// Fused attention: one block per (b,h). 8 waves; each wave owns 16 query rows
// per iteration (2 iterations cover N1=256).
//   k tile staged to LDS with gfx1250 async load-to-LDS (ASYNCcnt path);
//   S = qh*k^T (8 WMMAs) -> +bias+mask -> softmax (half-wave shuffles)
//   -> attn out (f32) + P in LDS (f16) -> P*V (8 WMMAs) -> xpre (f16)
// ---------------------------------------------------------------------------
__global__ __launch_bounds__(256) void attn_kernel(
    const _Float16* __restrict__ qh, const _Float16* __restrict__ kh,
    const _Float16* __restrict__ vh, const float* __restrict__ biasT,
    const float* __restrict__ mask, float* __restrict__ attn_out,
    _Float16* __restrict__ xpre) {
  __shared__ __align__(32) _Float16 kbuf[128 * 32];   // k tile: [n2][d], 8 KB
  __shared__ __align__(32) _Float16 vT[32 * 128];     // v^T: [d][n2],   8 KB
  __shared__ __align__(32) _Float16 pbuf[8][16 * 128];// per-wave probs, 32 KB

  const int tid  = threadIdx.x;
  const int lane = tid & 31;
  const int wave = tid >> 5;
  const int bh   = blockIdx.x;               // b*4+h
  const int b    = bh >> 2;
  const int h    = bh & 3;

  // ---- async-copy k tile (128x32 f16) into LDS: 512 x 16B segments -------
  // LDS address is taken from the vdst VGPR; generic->LDS offset is the low 32 bits.
#pragma unroll
  for (int it = 0; it < 2; ++it) {
    const int s   = it * 256 + tid;          // 0..511
    const int n2  = s >> 2;
    const int sub = s & 3;                   // 16B chunk within 64B row
    const _Float16* gp = kh + (b * 128 + n2) * 128 + h * 32 + sub * 8;
    unsigned ldsOff = (unsigned)(unsigned long long)(&kbuf[n2 * 32 + sub * 8]);
    asm volatile("global_load_async_to_lds_b128 %0, %1, off"
                 :: "v"(ldsOff), "v"(gp) : "memory");
  }

  // ---- stage v^T into LDS (transpose gather; coalesced global reads) ------
  for (int idx = tid; idx < 32 * 128; idx += 256) {
    const int d = idx & 31, n2 = idx >> 5;
    vT[d * 128 + n2] = vh[(b * 128 + n2) * 128 + h * 32 + d];
  }
  // prefetch the 2nd m-iteration's qh fragment rows
  __builtin_prefetch(qh + (b * 256 + (8 + wave) * 16) * 128 + h * 32, 0, 3);

  asm volatile("s_wait_asynccnt 0" ::: "memory");
  __syncthreads();

  const float* __restrict__ mbase = mask + (b & 63) * 32768;
  const float* __restrict__ bbase = biasT + h * 32768;
  float*       __restrict__ aout  = attn_out + bh * 32768;

  const int col16 = lane & 15;
  const int rAdd  = (lane >> 4) << 3;

#pragma unroll 1
  for (int mi = 0; mi < 2; ++mi) {
    const int mb = (mi * 8 + wave) * 16;     // first of 16 query rows

    // ---- S = qh * k^T : K-dim = hd = 32 -> one WMMA per 16x16 tile ----
    const v16h aq = a_frag_f16(qh + (b * 256 + mb) * 128 + h * 32, 128, lane);
    v8f acc[8];
#pragma unroll
    for (int nt = 0; nt < 8; ++nt) {
      v16h bk = b_frag_f16(kbuf + nt * 16 * 32, 32, lane);
      v8f z = {};
      acc[nt] = wmma16(aq, bk, z);
    }

    // ---- + relative-position bias + window mask ----
#pragma unroll
    for (int nt = 0; nt < 8; ++nt) {
      const int n2 = nt * 16 + col16;
#pragma unroll
      for (int r = 0; r < 8; ++r) {
        const int n1 = mb + rAdd + r;
        acc[nt][r] += bbase[n1 * 128 + n2] + mbase[n1 * 128 + n2];
      }
    }

    // ---- softmax over n2 (row lives in one 16-lane half of the wave) ----
    float inv[8];
#pragma unroll
    for (int r = 0; r < 8; ++r) {
      float mx = acc[0][r];
#pragma unroll
      for (int nt = 1; nt < 8; ++nt) mx = fmaxf(mx, acc[nt][r]);
      mx = fmaxf(mx, __shfl_xor(mx, 1, 32));
      mx = fmaxf(mx, __shfl_xor(mx, 2, 32));
      mx = fmaxf(mx, __shfl_xor(mx, 4, 32));
      mx = fmaxf(mx, __shfl_xor(mx, 8, 32));
      float s = 0.f;
#pragma unroll
      for (int nt = 0; nt < 8; ++nt) {
        const float e = __expf(acc[nt][r] - mx);
        acc[nt][r] = e;
        s += e;
      }
      s += __shfl_xor(s, 1, 32);
      s += __shfl_xor(s, 2, 32);
      s += __shfl_xor(s, 4, 32);
      s += __shfl_xor(s, 8, 32);
      inv[r] = 1.0f / s;
    }

    // ---- write attn probs (f32 output) + stage P (f16) for P*V ----
    _Float16* __restrict__ prow = pbuf[wave];
#pragma unroll
    for (int nt = 0; nt < 8; ++nt) {
      const int n2 = nt * 16 + col16;
#pragma unroll
      for (int r = 0; r < 8; ++r) {
        const float p = acc[nt][r] * inv[r];
        aout[(mb + rAdd + r) * 128 + n2] = p;
        prow[(rAdd + r) * 128 + n2] = (_Float16)p;
      }
    }
    __syncthreads();   // order P writes vs. fragment reads (uniform loop)

    // ---- xh = P * V : (16x128)*(128x32) = 2 n-tiles x 4 k-steps ----
#pragma unroll
    for (int nt = 0; nt < 2; ++nt) {
      v8f c = {};
#pragma unroll
      for (int ks = 0; ks < 4; ++ks) {
        v16h ap = a_frag_f16(prow + ks * 32, 128, lane);
        v16h bv = b_frag_f16(vT + nt * 16 * 128 + ks * 32, 128, lane);
        c = wmma16(ap, bv, c);
      }
      const int d = nt * 16 + col16;
#pragma unroll
      for (int r = 0; r < 8; ++r)
        xpre[(b * 256 + mb + rAdd + r) * 128 + h * 32 + d] = (_Float16)c[r];
    }
    __syncthreads();
  }
}

// ---------------------------------------------------------------------------
// proj: x = xpre @ proj_w^T + proj_b, f32 output
// ---------------------------------------------------------------------------
__global__ __launch_bounds__(256) void proj_kernel(
    const _Float16* __restrict__ xpre, const _Float16* __restrict__ wh,
    const float* __restrict__ pb, float* __restrict__ out) {
  const int lane = threadIdx.x & 31;
  const int wave = threadIdx.x >> 5;
  const int row0 = (blockIdx.x * 8 + wave) * 16;   // 16384 m-tiles

  v16h a[4];
#pragma unroll
  for (int ks = 0; ks < 4; ++ks)
    a[ks] = a_frag_f16(xpre + row0 * 128 + ks * 32, 128, lane);

  const int col16 = lane & 15;
  const int rAdd  = (lane >> 4) << 3;
#pragma unroll
  for (int nt = 0; nt < 8; ++nt) {
    v8f c = {};
#pragma unroll
    for (int ks = 0; ks < 4; ++ks) {
      v16h b = b_frag_f16(wh + (nt * 16) * 128 + ks * 32, 128, lane);
      c = wmma16(a[ks], b, c);
    }
    const float bv = pb[nt * 16 + col16];
    const int col = nt * 16 + col16;
#pragma unroll
    for (int r = 0; r < 8; ++r)
      out[(row0 + rAdd + r) * 128 + col] = c[r] + bv;
  }
}

// ---------------------------------------------------------------------------
// launch
// ---------------------------------------------------------------------------
extern "C" void kernel_launch(void* const* d_in, const int* in_sizes, int n_in,
                              void* d_out, int out_size, void* d_ws, size_t ws_size,
                              hipStream_t stream) {
  (void)in_sizes; (void)n_in; (void)out_size; (void)ws_size;

  const float* q    = (const float*)d_in[0];
  const float* kv   = (const float*)d_in[1];
  const float* mask = (const float*)d_in[2];
  const float* q_w  = (const float*)d_in[3];
  const float* q_b  = (const float*)d_in[4];
  const float* kv_w = (const float*)d_in[5];
  const float* kv_b = (const float*)d_in[6];
  const float* p_w  = (const float*)d_in[7];
  const float* p_b  = (const float*)d_in[8];
  const float* rpb  = (const float*)d_in[9];
  const int*   ridx = (const int*)d_in[10];

  // workspace layout (bytes); total ~192.6 MB
  char* ws = (char*)d_ws;
  _Float16* qh    = (_Float16*)(ws);                 // 64 MB  (B*N1*128 f16)
  _Float16* kh    = (_Float16*)(ws + 67108864ull);   // 32 MB  (B*N2*128 f16)
  _Float16* vh    = (_Float16*)(ws + 100663296ull);  // 32 MB
  _Float16* xpre  = (_Float16*)(ws + 134217728ull);  // 64 MB
  _Float16* qwh   = (_Float16*)(ws + 201326592ull);  // 32 KB
  _Float16* kvwh  = (_Float16*)(ws + 201359360ull);  // 64 KB
  _Float16* pwh   = (_Float16*)(ws + 201424896ull);  // 32 KB
  float*    biasT = (float*)   (ws + 201457664ull);  // 512 KB (NH*N1*N2 f32)

  float* xout = (float*)d_out;                       // (B,N1,C)
  float* aout = xout + 33554432ull;                  // (B,NH,N1,N2)

  prep_kernel  <<<512,  256, 0, stream>>>(q_w, kv_w, p_w, rpb, ridx,
                                          qwh, kvwh, pwh, biasT);
  qproj_kernel <<<2048, 256, 0, stream>>>(q, qwh, q_b, qh);
  kvproj_kernel<<<1024, 256, 0, stream>>>(kv, kvwh, kv_b, kh, vh);
  attn_kernel  <<<4096, 256, 0, stream>>>(qh, kh, vh, biasT, mask, aout, xpre);
  proj_kernel  <<<2048, 256, 0, stream>>>(xpre, pwh, p_b, xout);
}
